// AsyncConvBis_50019189129835
// MI455X (gfx1250) — compile-verified, matrix operand order checked by reading, not run on hardware
//
#include <hip/hip_runtime.h>

// Problem constants (from reference)
#define NVV   25000
#define BBB   2
#define CCH   32
#define NRR   3
#define NDD   16
#define FFF   64
#define TOTV  (BBB*NVV)        // 50000 vertices
#define KTOT  1568             // NR*ND*C + C (center folded in)
#define KSTEPS 49              // 1568 / 32
#define AROW  1576             // padded LDS row stride in halfs (bank spread)
#define MB    64               // vertices per block (4 M-tiles)

typedef __attribute__((ext_vector_type(16))) _Float16     v16h;
typedef __attribute__((ext_vector_type(8)))  float        v8f;
typedef __attribute__((ext_vector_type(4)))  float        f32x4;
typedef __attribute__((ext_vector_type(4)))  _Float16     h16x4;
typedef __attribute__((ext_vector_type(4)))  unsigned int u32x4;
typedef __attribute__((ext_vector_type(8)))  unsigned int u32x8;

static __device__ __forceinline__ v16h to_v16h(u32x4 a, u32x4 b) {
  union { u32x4 q[2]; v16h v; } u; u.q[0] = a; u.q[1] = b; return u.v;
}
static __device__ __forceinline__ v16h to_v16h8(u32x8 a) {
  union { u32x8 q; v16h v; } u; u.q = a; return u.v;
}

// ---------------------------------------------------------------------------
// Kernel 1: convert y (f32) -> f16 once.  1,600,000 elems = 400,000 vec4.
// ---------------------------------------------------------------------------
__global__ __launch_bounds__(256) void cvt_y(const float* __restrict__ y,
                                             _Float16* __restrict__ yh, int n4) {
  int i = blockIdx.x * 256 + threadIdx.x;
  if (i < n4) {
    f32x4 v = ((const f32x4*)y)[i];
    h16x4 o;
    o.x = (_Float16)v.x; o.y = (_Float16)v.y;
    o.z = (_Float16)v.z; o.w = (_Float16)v.w;
    ((h16x4*)yh)[i] = o;
  }
}

// ---------------------------------------------------------------------------
// Kernel 2: 16 rotated weight matrices (+center_kernel rows) in f16, stored
// pre-swizzled in the wave32 WMMA B-fragment lane layout:
//   Bp[((d*4+ft)*49+ks)*512 + lane*16 + h]
// ---------------------------------------------------------------------------
__global__ __launch_bounds__(256) void pack_B(const float* __restrict__ ker,
                                              const float* __restrict__ ck,
                                              _Float16* __restrict__ Bp) {
  int idx  = blockIdx.x * 256 + threadIdx.x;   // exactly 1,605,632 launched
  int h    = idx & 15;
  int slot = idx >> 4;
  int lane = slot & 31;
  int t    = slot >> 5;
  int ks   = t % KSTEPS;
  int t2   = t / KSTEPS;
  int ft   = t2 & 3;
  int d    = t2 >> 2;                          // rotation 0..15
  int f    = ft * 16 + (lane & 15);            // output feature 0..63
  int hi   = lane >> 4;
  int kh   = (h < 8) ? (hi * 8 + h) : (16 + hi * 8 + (h - 8));
  int kk   = ks * 32 + kh;                     // K index 0..1567
  float wv;
  if (kk < 1536) {
    int r   = kk >> 9;        // / (ND*C)
    int rem = kk & 511;
    int k   = rem >> 5;
    int c   = rem & 31;
    int j   = (k - d + 16) & 15;               // rotated kernel tap
    wv = ker[(((r * 16) + j) * 32 + c) * 64 + f];
  } else {
    int c = kk - 1536;                         // center-kernel rows
    wv = ck[c * 64 + f];
  }
  Bp[idx] = (_Float16)wv;
}

// ---------------------------------------------------------------------------
// Kernel 3: main GEMM + max/ReLU epilogue.
// Block = 64 vertices (4 M-tiles) x 8 waves.  Wave w: mt=w&3, half h=w>>2.
// Outer c=0..3 over direction quadruples so all waves of a half read the SAME
// B fragments in lock-step (WGP$ reuse). A+B double-buffered in registers.
// ---------------------------------------------------------------------------
__global__ __launch_bounds__(256) void conv_main(const _Float16* __restrict__ yh,
                                                 const int* __restrict__ em,
                                                 const _Float16* __restrict__ Bp,
                                                 const float* __restrict__ bias,
                                                 float* __restrict__ out) {
  __shared__ __align__(16) _Float16 As[MB * AROW];   // 201,728 bytes

  const int tid   = threadIdx.x;
  const int gbase = blockIdx.x * MB;

  // ---- stage gathered A tile (64 x 1568 f16) into LDS ----
  for (int idx = tid; idx < MB * KSTEPS; idx += 256) {   // 3136 chunks
    int m = idx / KSTEPS, q = idx - m * KSTEPS;
    int g = gbase + m;
    if (g >= TOTV) g = TOTV - 1;       // tail block: harmless duplicate rows
    int b = g / NVV, v = g - b * NVV;
    int src;
    if (q < 48) {
      int r = q >> 4, k = q & 15;
      size_t emi = ((((size_t)b * NVV + v) * NRR + r) * NDD + k) * 2;
      int bb = em[emi];
      int vv = em[emi + 1];
      src = bb * NVV + vv;
    } else {
      src = g;                         // center-vertex channels (K 1536..1567)
    }
    const u32x4* sp = (const u32x4*)(yh + (size_t)src * CCH);
    u32x4* dp = (u32x4*)(As + m * AROW + q * CCH);
    dp[0] = sp[0]; dp[1] = sp[1]; dp[2] = sp[2]; dp[3] = sp[3];
  }
  __syncthreads();

  const int lane = tid & 31;
  const int w    = tid >> 5;
  const int mt   = w & 3;              // M tile 0..3
  const int h    = w >> 2;             // direction half 0..1
  const int hi   = lane >> 4;
  const _Float16* arow = As + (mt * 16 + (lane & 15)) * AROW + hi * 8;

  v8f vmax[4];
#pragma unroll
  for (int ft = 0; ft < 4; ++ft)
#pragma unroll
    for (int e = 0; e < 8; ++e) vmax[ft][e] = -3.0e38f;

  struct AFrag { u32x4 q0, q1; };
  struct BFrag { u32x8 u; };

#define LOADA(A, ks) do {                                            \
    (A).q0 = *(const u32x4*)(arow + (ks) * 32);                      \
    (A).q1 = *(const u32x4*)(arow + (ks) * 32 + 16);                 \
  } while (0)

#define LOADB(Bf, ks) do {                                           \
    _Pragma("unroll")                                                \
    for (int j = 0; j < 8; ++j)                                      \
      (Bf)[j].u = *(const u32x8*)(Bp + boff[j] + (size_t)(ks) * 512);\
  } while (0)

#define COMP(A, Bf) do {                                             \
    v16h av = to_v16h((A).q0, (A).q1);                               \
    _Pragma("unroll")                                                \
    for (int j = 0; j < 8; ++j)                                      \
      acc[j] = __builtin_amdgcn_wmma_f32_16x16x32_f16(               \
          false, av, false, to_v16h8((Bf)[j].u), (short)0,           \
          acc[j], false, false);                                     \
  } while (0)

  for (int c = 0; c < 4; ++c) {
    v8f acc[8];
#pragma unroll
    for (int j = 0; j < 8; ++j)
#pragma unroll
      for (int e = 0; e < 8; ++e) acc[j][e] = 0.f;

    size_t boff[8];
#pragma unroll
    for (int j = 0; j < 8; ++j) {
      int dd = j >> 2, ft = j & 3;
      int d  = c * 4 + h * 2 + dd;
      boff[j] = ((size_t)(d * 4 + ft) * KSTEPS) * 512 + (size_t)lane * 16;
    }

    AFrag a0, a1;
    BFrag b0[8], b1[8];
    LOADA(a0, 0); LOADB(b0, 0);
#pragma unroll 1
    for (int ks = 0; ks < 48; ks += 2) {
      LOADA(a1, ks + 1); LOADB(b1, ks + 1);
      COMP(a0, b0);
      LOADA(a0, ks + 2); LOADB(b0, ks + 2);   // ks+2 <= 48
      COMP(a1, b1);
    }
    COMP(a0, b0);                              // ks = 48

    // fold this c's two directions into the per-ft running max
#pragma unroll
    for (int ft = 0; ft < 4; ++ft)
#pragma unroll
      for (int e = 0; e < 8; ++e)
        vmax[ft][e] = fmaxf(vmax[ft][e],
                            fmaxf(acc[ft][e], acc[4 + ft][e]));
  }

  __syncthreads();                     // all LDS A reads finished; reuse LDS
  float* red = (float*)As;             // 8 waves * 4 ft * 256 floats = 32 KB
  const int mbse = hi * 8;
  const int nn = lane & 15;
#pragma unroll
  for (int ft = 0; ft < 4; ++ft)
#pragma unroll
    for (int e = 0; e < 8; ++e)
      red[(w * 4 + ft) * 256 + (mbse + e) * 16 + nn] = vmax[ft][e];
  __syncthreads();

  // ---- combine halves h=0 (w=mt) and h=1 (w=4+mt), +bias, ReLU, store ----
  for (int o = tid; o < MB * FFF; o += 256) {  // 4096 outputs
    int m   = o >> 6, f = o & 63;
    int mt2 = m >> 4, ml = m & 15;
    int ft2 = f >> 4,  c2 = f & 15;
    float x0 = red[(mt2 * 4 + ft2) * 256 + ml * 16 + c2];
    float x1 = red[((4 + mt2) * 4 + ft2) * 256 + ml * 16 + c2];
    float val = fmaxf(x0, x1) + bias[f];
    val = fmaxf(val, 0.f);
    int g = gbase + m;
    if (g < TOTV) out[(size_t)g * FFF + f] = val;
  }
#undef LOADA
#undef LOADB
#undef COMP
}

// ---------------------------------------------------------------------------
extern "C" void kernel_launch(void* const* d_in, const int* in_sizes, int n_in,
                              void* d_out, int out_size, void* d_ws, size_t ws_size,
                              hipStream_t stream) {
  const float* y    = (const float*)d_in[0];
  const int*   em   = (const int*)d_in[1];
  const float* ker  = (const float*)d_in[2];
  const float* ck   = (const float*)d_in[3];
  const float* bias = (const float*)d_in[4];
  float* out = (float*)d_out;

  _Float16* yh = (_Float16*)d_ws;                              // 3,200,000 B
  _Float16* Bp = (_Float16*)((char*)d_ws + 3200000);           // 3,211,264 B

  int n4 = (BBB * NVV * CCH) / 4;                              // 400,000
  cvt_y<<<(n4 + 255) / 256, 256, 0, stream>>>(y, yh, n4);
  pack_B<<<6272, 256, 0, stream>>>(ker, ck, Bp);               // 1,605,632 elems
  int nblk = (TOTV + MB - 1) / MB;                             // 782
  conv_main<<<nblk, 256, 0, stream>>>(yh, em, Bp, bias, out);
}
